// _SEAggregation_71511205478484
// MI455X (gfx1250) — matrix operand order repaired and checked by reading, not compiled, not run on hardware
//
#include <hip/hip_runtime.h>
#include <hip/hip_bf16.h>
#include <math.h>

// Problem constants (match reference)
#define IN_DIM  512
#define OUT_DIM 256
#define KHOPS   4
// E_CHS == KHOPS+1 == 5

typedef __attribute__((ext_vector_type(16))) __bf16 v16bf;
typedef __attribute__((ext_vector_type(8)))  __bf16 v8bf;
typedef __attribute__((ext_vector_type(4)))  __bf16 v4bf;
typedef __attribute__((ext_vector_type(8)))  float  v8f;

// float -> bf16 with round-to-nearest-even (used only in one-shot prep kernels)
__device__ __forceinline__ __bf16 f2bf(float f) {
    unsigned u = __builtin_bit_cast(unsigned, f);
    unsigned r = u + 0x7FFFu + ((u >> 16) & 1u);
    unsigned short h = (unsigned short)(r >> 16);
    return __builtin_bit_cast(__bf16, h);
}

// ---------------- zero / degrees / norms ----------------

__global__ void k_zero(float* __restrict__ p, int n) {
    int i = blockIdx.x * 256 + threadIdx.x;
    if (i < n) p[i] = 0.0f;
}

__global__ void k_degree(const int* __restrict__ src, const int* __restrict__ dst,
                         float* __restrict__ degO, float* __restrict__ degI, int nE) {
    int i = blockIdx.x * 256 + threadIdx.x;
    if (i < nE) {
        atomicAdd(&degO[src[i]], 1.0f);
        atomicAdd(&degI[dst[i]], 1.0f);
    }
}

// in-place: deg -> deg^{-1/2} with clamp(deg,1)
__global__ void k_norm(float* __restrict__ degO, float* __restrict__ degI, int nN) {
    int i = blockIdx.x * 256 + threadIdx.x;
    if (i < nN) {
        degO[i] = rsqrtf(fmaxf(degO[i], 1.0f));
        degI[i] = rsqrtf(fmaxf(degI[i], 1.0f));
    }
}

// ---------------- GEMM prep: one-shot conversions ----------------

// A (f32 [nN,512]) -> Abf (bf16 [nN,512]), 4 elements/thread
__global__ void k_cvtA(const float* __restrict__ A, __bf16* __restrict__ Ab, int n4) {
    int i = blockIdx.x * 256 + threadIdx.x;
    if (i < n4) {
        float4 f = ((const float4*)A)[i];
        v4bf o;
        o[0] = f2bf(f.x); o[1] = f2bf(f.y); o[2] = f2bf(f.z); o[3] = f2bf(f.w);
        ((v4bf*)Ab)[i] = o;
    }
}

// Pack weight [512,256] f32 into per-lane WMMA B fragments (bf16):
//   Bpack[((ntile*16 + kb16)*32 + lane)*16 + e] = bf16(W[(kb16*32 + half*16 + e)*256 + ntile*16 + l16])
// (B fragment in contiguous K order: lane half selects K-half, elem e -> K = half*16+e, lane&15 = N)
__global__ void k_packB(const float* __restrict__ W, __bf16* __restrict__ Bp) {
    int gid = blockIdx.x * 256 + threadIdx.x;      // 16*16*32*16 = 131072 total
    if (gid >= 16 * 16 * 32 * 16) return;
    int e     = gid & 15;
    int lane  = (gid >> 4) & 31;
    int kb16  = (gid >> 9) & 15;
    int ntile = gid >> 13;
    int half  = lane >> 4;
    int l16   = lane & 15;
    int k     = kb16 * 32 + half * 16 + e;
    Bp[gid] = f2bf(W[(size_t)k * OUT_DIM + ntile * 16 + l16]);
}

// ---------------- GEMM: H = Abf @ Bpack, f32 accumulate ----------------
// One wave per 16x64 output strip: 1 A-fragment load feeds 4 WMMAs per K-step.
// All four B fragment pairs are loaded into distinct registers before the WMMA
// sequence so the loads issue as one clause and the waits stage down
// (loadcnt 6/4/2/0) instead of serializing load->wmma.
__global__ void __launch_bounds__(256)
k_gemm(const __bf16* __restrict__ Abf, const __bf16* __restrict__ Bpack,
       float* __restrict__ H, int nN) {
    const int lane = threadIdx.x & 31;
    const int wave = threadIdx.x >> 5;
    const int half = lane >> 4;
    const int l16  = lane & 15;

    const int wtile = blockIdx.x * 8 + wave;   // wave strip index
    const int mtile = wtile >> 2;              // OUT_DIM/64 = 4 col groups
    const int cg    = wtile & 3;
    const int row0  = mtile * 16;
    if (row0 >= nN) return;                    // nN is a multiple of 16 (50000)

    const __bf16* Arow = Abf + (size_t)(row0 + l16) * IN_DIM;
    // lane-fragment base for this col group: ntile = cg*4 + j
    const __bf16* Bbase = Bpack + ((size_t)(cg * 4) * 16 * 32 + lane) * 16;

    v8f c[4] = {{}, {}, {}, {}};
    #pragma unroll 4
    for (int kb16 = 0; kb16 < 16; ++kb16) {
        const int kb = kb16 * 32;
        // A fragment: two contiguous 8-element bf16 runs (contiguous K order, K = half*16 + e, as in Bpack)
        v8bf alo = *(const v8bf*)(Arow + kb + half * 16);
        v8bf ahi = *(const v8bf*)(Arow + kb + half * 16 + 8);

        // Issue all four B fragment pairs up front (distinct registers)
        v8bf bl[4], bh[4];
        #pragma unroll
        for (int j = 0; j < 4; ++j) {
            const __bf16* bp = Bbase + ((size_t)j * 16 + kb16) * 32 * 16;
            bl[j] = *(const v8bf*)(bp);
            bh[j] = *(const v8bf*)(bp + 8);
        }

        v16bf a;
        #pragma unroll
        for (int e = 0; e < 8; ++e) { a[e] = alo[e]; a[8 + e] = ahi[e]; }

        #pragma unroll
        for (int j = 0; j < 4; ++j) {
            v16bf b;
            #pragma unroll
            for (int e = 0; e < 8; ++e) { b[e] = bl[j][e]; b[8 + e] = bh[j][e]; }
            c[j] = __builtin_amdgcn_wmma_f32_16x16x32_bf16(
                       false, a, false, b, (short)0, c[j], false, false);
        }
    }

    // f32 C/D 16x16 layout: vgpr r -> M = half*8 + r, lane&15 = N
    #pragma unroll
    for (int j = 0; j < 4; ++j) {
        float* Hrow = H + (size_t)(row0 + half * 8) * OUT_DIM + (cg * 4 + j) * 16 + l16;
        #pragma unroll
        for (int r = 0; r < 8; ++r)
            Hrow[(size_t)r * OUT_DIM] = c[j][r];
    }
}

// ---------------- hop propagation ----------------

// xs = hop_prev * norm_out[node];  acc = 0
__global__ void k_prep(const float* __restrict__ hop_prev, const float* __restrict__ normO,
                       float* __restrict__ xs, float* __restrict__ acc, int total) {
    int i = blockIdx.x * 256 + threadIdx.x;
    if (i < total) {
        xs[i]  = hop_prev[i] * normO[i >> 8];   // OUT_DIM == 256
        acc[i] = 0.0f;
    }
}

// acc[dst] += xs[src]   (edge index uniform across block -> scalar src/dst loads,
// lane = channel -> coalesced gather + coalesced f32 atomics, all L2-resident)
__global__ void __launch_bounds__(256)
k_scatter(const float* __restrict__ xs, const int* __restrict__ src,
          const int* __restrict__ dst, float* __restrict__ acc, int nE) {
    const int ch = threadIdx.x;          // 0..255 channel
    const int e0 = blockIdx.x * 8;
    #pragma unroll
    for (int i = 0; i < 8; ++i) {
        int e = e0 + i;
        if (e >= nE) return;
        int s = src[e];
        int d = dst[e];
        if (e + 1 < nE)                  // pull next source row toward L2/L0
            __builtin_prefetch(&xs[(size_t)src[e + 1] * OUT_DIM + ch], 0, 1);
        atomicAdd(&acc[(size_t)d * OUT_DIM + ch], xs[(size_t)s * OUT_DIM + ch]);
    }
}

// acc *= norm_in[node]
__global__ void k_scale_in(float* __restrict__ acc, const float* __restrict__ normI, int total) {
    int i = blockIdx.x * 256 + threadIdx.x;
    if (i < total) acc[i] *= normI[i >> 8];
}

// ---------------- fused squeeze-excitation epilogue ----------------
// One wave32 per node. Lanes hold 8 channels x 5 hops in registers.
__global__ void __launch_bounds__(256)
k_epilogue(const float* __restrict__ hops, const float* __restrict__ e_att,
           const float* __restrict__ W1, const float* __restrict__ W2,
           float* __restrict__ out, int nN) {
    const int lane = threadIdx.x & 31;
    const int node = blockIdx.x * 8 + (threadIdx.x >> 5);
    if (node >= nN) return;
    const size_t hopStride = (size_t)nN * OUT_DIM;

    float att[8];
    #pragma unroll
    for (int j = 0; j < 8; ++j) att[j] = e_att[lane + 32 * j];

    float hv[KHOPS + 1][8];
    float sq[KHOPS + 1];
    #pragma unroll
    for (int k = 0; k <= KHOPS; ++k) {
        const float* hp = hops + (size_t)k * hopStride + (size_t)node * OUT_DIM;
        float acc = 0.0f;
        #pragma unroll
        for (int j = 0; j < 8; ++j) {
            float v = hp[lane + 32 * j];
            hv[k][j] = v;
            acc += v * att[j];
        }
        #pragma unroll
        for (int off = 16; off > 0; off >>= 1)
            acc += __shfl_xor(acc, off, 32);
        sq[k] = acc;
    }

    float nrm2 = 0.0f;
    #pragma unroll
    for (int k = 0; k <= KHOPS; ++k) nrm2 += sq[k] * sq[k];
    float inv = 1.0f / fmaxf(sqrtf(nrm2), 1e-12f);

    float e1[KHOPS + 1];
    #pragma unroll
    for (int j = 0; j <= KHOPS; ++j) {
        float s = 0.0f;
        #pragma unroll
        for (int k = 0; k <= KHOPS; ++k) s += sq[k] * inv * W1[k * (KHOPS + 1) + j];
        e1[j] = fminf(fmaxf(s, 0.0f), 6.0f);
    }
    float g[KHOPS + 1];
    #pragma unroll
    for (int k = 0; k <= KHOPS; ++k) {
        float s = 0.0f;
        #pragma unroll
        for (int c = 0; c <= KHOPS; ++c) s += e1[c] * W2[c * (KHOPS + 1) + k];
        g[k] = tanhf(s);
    }

    float* op = out + (size_t)node * OUT_DIM;
    #pragma unroll
    for (int j = 0; j < 8; ++j) {
        float s = 0.0f;
        #pragma unroll
        for (int k = 0; k <= KHOPS; ++k) s += hv[k][j] * g[k];
        op[lane + 32 * j] = s;
    }
}

// ---------------- launch ----------------

extern "C" void kernel_launch(void* const* d_in, const int* in_sizes, int n_in,
                              void* d_out, int out_size, void* d_ws, size_t ws_size,
                              hipStream_t stream) {
    const float* n_feat = (const float*)d_in[0];
    const float* weight = (const float*)d_in[1];
    const float* w1     = (const float*)d_in[2];
    const float* w2     = (const float*)d_in[3];
    const float* e_att  = (const float*)d_in[4];
    const int*   src    = (const int*)d_in[5];
    const int*   dst    = (const int*)d_in[6];
    float*       out    = (float*)d_out;

    const int nN = in_sizes[0] / IN_DIM;   // 50000
    const int nE = in_sizes[5];            // 800000
    const int total = nN * OUT_DIM;        // 12.8M

    // Workspace layout (floats):
    //   hops[5][nN*256] | xs[nN*256] | degO[nN] | degI[nN] | Bpack[128K bf16]
    // Abf (bf16 [nN,512], 51.2MB) aliases xs: it is dead once the GEMM finishes,
    // and xs is first written after the GEMM.
    float* ws   = (float*)d_ws;
    size_t hopStride = (size_t)nN * OUT_DIM;
    float*  hops  = ws;
    float*  xs    = hops + 5 * hopStride;
    __bf16* Abf   = (__bf16*)xs;
    float*  degO  = xs + hopStride;
    float*  degI  = degO + nN;
    __bf16* Bpack = (__bf16*)(degI + nN);

    // degrees -> norms (must re-zero every call: atomically accumulated)
    k_zero<<<(2 * nN + 255) / 256, 256, 0, stream>>>(degO, 2 * nN);
    k_degree<<<(nE + 255) / 256, 256, 0, stream>>>(src, dst, degO, degI, nE);
    k_norm<<<(nN + 255) / 256, 256, 0, stream>>>(degO, degI, nN);

    // one-shot bf16 conversions (off the GEMM critical path)
    {
        int n4 = nN * IN_DIM / 4;
        k_cvtA<<<(n4 + 255) / 256, 256, 0, stream>>>(n_feat, Abf, n4);
        k_packB<<<(16 * 16 * 32 * 16 + 255) / 256, 256, 0, stream>>>(weight, Bpack);
    }

    // hop0 = n_feat @ weight (bf16 WMMA, f32 accumulate); 16x64 strip per wave
    {
        int mtiles = (nN + 15) / 16;
        int wtiles = mtiles * (OUT_DIM / 64);
        k_gemm<<<(wtiles + 7) / 8, 256, 0, stream>>>(Abf, Bpack, hops, nN);
    }

    // 4 graph-conv hops: hop_k = D_in^{-1/2} A D_out^{-1/2} hop_{k-1}
    for (int k = 1; k <= KHOPS; ++k) {
        float* prev = hops + (size_t)(k - 1) * hopStride;
        float* cur  = hops + (size_t)k * hopStride;
        k_prep<<<(total + 255) / 256, 256, 0, stream>>>(prev, degO, xs, cur, total);
        k_scatter<<<(nE + 7) / 8, 256, 0, stream>>>(xs, src, dst, cur, nE);
        k_scale_in<<<(total + 255) / 256, 256, 0, stream>>>(cur, degI, total);
    }

    // fused squeeze-excitation + recombine
    k_epilogue<<<(nN + 7) / 8, 256, 0, stream>>>(hops, e_att, w1, w2, out, nN);
}
